// MultiModalFusionGAT_78743930405084
// MI455X (gfx1250) — matrix operand order repaired and checked by reading, not compiled, not run on hardware
//
#include <hip/hip_runtime.h>
#include <hip/hip_bf16.h>
#include <stdint.h>

typedef __bf16 bf16;
typedef __attribute__((ext_vector_type(16))) __bf16 v16bf;
typedef __attribute__((ext_vector_type(8)))  float  v8f;

// ---------------------------------------------------------------------------
// WMMA helpers (wave32, v_wmma_f32_16x16x32_bf16)
// ---------------------------------------------------------------------------
__device__ __forceinline__ v8f wmma_bf16(v16bf a, v16bf b, v8f c) {
  return __builtin_amdgcn_wmma_f32_16x16x32_bf16(
      /*neg_a=*/false, a, /*neg_b=*/false, b,
      /*c_mod=*/(short)0, c, /*reuse_a=*/false, /*reuse_b=*/false);
}

// LDS tile geometry per buffer: A: 128 rows x 32 K (stride 32, bf16)
//                               B:  64 cols x 32 K transposed (stride 32)
__device__ __forceinline__ v16bf ldsA_frag(const bf16* As, int rowBase, int lane) {
  const int m16 = lane & 15, half = lane >> 4;
  v16bf af;
  const bf16* arow = As + (rowBase + m16) * 32 + 8 * half;
  #pragma unroll
  for (int j = 0; j < 16; ++j) {
    const int g = j >> 1, p = j & 1;
    af[j] = arow[(g < 4 ? 2 * g : 16 + 2 * (g - 4)) + p];
  }
  return af;
}
__device__ __forceinline__ v16bf ldsB_frag(const bf16* Bs, int colBase, int lane) {
  const int n16 = lane & 15, half = lane >> 4;
  v16bf bfv;
  const bf16* brow = Bs + (colBase + n16) * 32 + 8 * half;
  #pragma unroll
  for (int j = 0; j < 16; ++j) bfv[j] = brow[j < 8 ? j : 8 + j];
  return bfv;
}

// 4 WMMAs per wave per K-step: wave tile 32(M) x 32(N)
__device__ __forceinline__ void mma_step(const bf16* Abuf, const bf16* Bbuf,
                                         int wm, int wn, int lane,
                                         v8f& a00, v8f& a01, v8f& a10, v8f& a11) {
  const v16bf af0 = ldsA_frag(Abuf, wm * 32, lane);
  const v16bf af1 = ldsA_frag(Abuf, wm * 32 + 16, lane);
  const v16bf bf0 = ldsB_frag(Bbuf, wn * 32, lane);
  const v16bf bf1 = ldsB_frag(Bbuf, wn * 32 + 16, lane);
  a00 = wmma_bf16(af0, bf0, a00);
  a01 = wmma_bf16(af0, bf1, a01);
  a10 = wmma_bf16(af1, bf0, a10);
  a11 = wmma_bf16(af1, bf1, a11);
}

// CDNA5 async global->LDS copy of 16 bytes (8 bf16), ASYNCcnt-tracked.
__device__ __forceinline__ void async_copy_b128(const bf16* gsrc, bf16* ldst) {
  const unsigned lds_addr = (unsigned)(uintptr_t)ldst;       // low 32 bits = LDS offset
  const unsigned long long gaddr = (unsigned long long)(uintptr_t)gsrc;
  asm volatile("global_load_async_to_lds_b128 %0, %1, off"
               :: "v"(lds_addr), "v"(gaddr) : "memory");
}
__device__ __forceinline__ void async_wait0() {
  asm volatile("s_wait_asynccnt 0x0" ::: "memory");
}

// ---------------------------------------------------------------------------
// Kernel 1: KQV projection  C[M x 768] = X[M x 256] @ W[256 x 768] + b
// Block 256 thr (8 waves) -> 128x64 tile; wave 32x32; double-buffered LDS.
// grid = (ceil(M/128), 12)
// ---------------------------------------------------------------------------
__global__ __launch_bounds__(256)
void kqv_gemm(const float* __restrict__ X, const float* __restrict__ W,
              const float* __restrict__ bias, int M,
              bf16* __restrict__ kOut, bf16* __restrict__ qOut,
              bf16* __restrict__ vOut) {
  __shared__ bf16 As[2][128 * 32];
  __shared__ bf16 Bs[2][64 * 32];
  const int tid = threadIdx.x, lane = tid & 31, wave = tid >> 5;
  const int wm = wave >> 1, wn = wave & 1;     // 4 x 2 waves of 32x32
  const int row0 = blockIdx.x * 128, col0 = blockIdx.y * 64;

  const int ar = tid >> 1, ac16 = (tid & 1) * 16;  // A: row ar, 16 ks
  const int bn = tid & 63, bk0 = (tid >> 6) * 8;   // B: col bn, 8 ks
  float4 arg[4];
  float breg[8];

  auto loadG = [&](int kb) {
    #pragma unroll
    for (int q = 0; q < 4; ++q) arg[q] = make_float4(0.f, 0.f, 0.f, 0.f);
    const int grow = row0 + ar;
    if (grow < M) {
      const float* p = X + (size_t)grow * 256 + kb + ac16;
      #pragma unroll
      for (int q = 0; q < 4; ++q) arg[q] = *(const float4*)(p + 4 * q);
    }
    #pragma unroll
    for (int u = 0; u < 8; ++u)
      breg[u] = W[(size_t)(kb + bk0 + u) * 768 + col0 + bn];
  };
  auto storeL = [&](int buf) {
    bf16* arow = &As[buf][ar * 32 + ac16];
    #pragma unroll
    for (int q = 0; q < 4; ++q) {
      arow[4 * q + 0] = (bf16)arg[q].x; arow[4 * q + 1] = (bf16)arg[q].y;
      arow[4 * q + 2] = (bf16)arg[q].z; arow[4 * q + 3] = (bf16)arg[q].w;
    }
    bf16* brow = &Bs[buf][bn * 32 + bk0];
    #pragma unroll
    for (int u = 0; u < 8; ++u) brow[u] = (bf16)breg[u];
  };

  v8f a00 = {}, a01 = {}, a10 = {}, a11 = {};
  loadG(0);
  storeL(0);
  __syncthreads();
  constexpr int S = 256 / 32;
  #pragma unroll
  for (int s = 0; s < S; ++s) {
    const int cur = s & 1;
    if (s + 1 < S) loadG((s + 1) * 32);
    mma_step(As[cur], Bs[cur], wm, wn, lane, a00, a01, a10, a11);
    if (s + 1 < S) storeL(1 - cur);
    __syncthreads();
  }

  const int nl = lane & 15, half = lane >> 4;
  const int colA = col0 + wn * 32 + nl;
  const int colB = colA + 16;
  const float biasA = bias[colA], biasB = bias[colB];
  auto put = [&](int m, int gcol, float val) {
    const int which = gcol >> 8, cc = gcol & 255;   // 0=k 1=q 2=v
    const bf16 o = (bf16)val;
    if (which == 0)      kOut[(size_t)m * 256 + cc] = o;
    else if (which == 1) { if (qOut) qOut[(size_t)m * 256 + cc] = o; }
    else                 vOut[(size_t)m * 256 + cc] = o;
  };
  #pragma unroll
  for (int r = 0; r < 8; ++r) {
    const int m0 = row0 + wm * 32 + r + 8 * half;
    const int m1 = m0 + 16;
    if (m0 < M) { put(m0, colA, a00[r] + biasA); put(m0, colB, a01[r] + biasB); }
    if (m1 < M) { put(m1, colA, a10[r] + biasA); put(m1, colB, a11[r] + biasB); }
  }
}

// ---------------------------------------------------------------------------
// Kernel 2: relation transform  out[n,h,:] = in[n,h,:] @ Wrel[h] (64x64)
// A operand is bf16 in memory -> async global->LDS copy (no staging regs).
// grid = (ceil(M/128), 4 heads)
// ---------------------------------------------------------------------------
__global__ __launch_bounds__(256)
void rel_gemm(const bf16* __restrict__ In, const float* __restrict__ Wrel,
              int M, int et, bf16* __restrict__ Out) {
  __shared__ bf16 As[2][128 * 32];
  __shared__ bf16 Bs[2][64 * 32];
  const int tid = threadIdx.x, lane = tid & 31, wave = tid >> 5;
  const int wm = wave >> 1, wn = wave & 1;
  const int row0 = blockIdx.x * 128;
  const int h = blockIdx.y;
  const float* Wh = Wrel + ((size_t)et * 4 + h) * 4096;

  const int bn = tid & 63, bk0 = (tid >> 6) * 8;
  float breg[8];

  auto asyncA = [&](int kb, int buf) {            // 128x32 bf16 = 512 b128 units
    #pragma unroll
    for (int q = 0; q < 2; ++q) {
      const int u = tid + q * 256;
      const int r = u >> 2, k8 = (u & 3) * 8;
      const int grow = row0 + r;
      if (grow < M)
        async_copy_b128(In + (size_t)grow * 256 + h * 64 + kb + k8,
                        &As[buf][r * 32 + k8]);
    }
  };
  auto loadB = [&](int kb) {
    #pragma unroll
    for (int u = 0; u < 8; ++u)
      breg[u] = Wh[(size_t)(kb + bk0 + u) * 64 + bn];
  };
  auto storeB = [&](int buf) {
    bf16* brow = &Bs[buf][bn * 32 + bk0];
    #pragma unroll
    for (int u = 0; u < 8; ++u) brow[u] = (bf16)breg[u];
  };

  v8f a00 = {}, a01 = {}, a10 = {}, a11 = {};
  asyncA(0, 0);
  loadB(0);
  storeB(0);
  async_wait0();
  __syncthreads();
  constexpr int S = 64 / 32;
  #pragma unroll
  for (int s = 0; s < S; ++s) {
    const int cur = s & 1;
    if (s + 1 < S) { asyncA((s + 1) * 32, 1 - cur); loadB((s + 1) * 32); }
    mma_step(As[cur], Bs[cur], wm, wn, lane, a00, a01, a10, a11);
    if (s + 1 < S) { storeB(1 - cur); async_wait0(); }
    __syncthreads();
  }

  const int nl = lane & 15, half = lane >> 4;
  const int colA = wn * 32 + nl;                 // 0..63
  #pragma unroll
  for (int r = 0; r < 8; ++r) {
    const int m0 = row0 + wm * 32 + r + 8 * half;
    const int m1 = m0 + 16;
    if (m0 < M) {
      Out[(size_t)m0 * 256 + h * 64 + colA]      = (bf16)a00[r];
      Out[(size_t)m0 * 256 + h * 64 + colA + 16] = (bf16)a01[r];
    }
    if (m1 < M) {
      Out[(size_t)m1 * 256 + h * 64 + colA]      = (bf16)a10[r];
      Out[(size_t)m1 * 256 + h * 64 + colA + 16] = (bf16)a11[r];
    }
  }
}

// ---------------------------------------------------------------------------
// Kernel 3: edge attention scores + segment max (ordered-uint atomicMax)
// ---------------------------------------------------------------------------
__device__ __forceinline__ unsigned int fmax_encode(float x) {
  const unsigned int u = __float_as_uint(x);
  return (u & 0x80000000u) ? ~u : (u | 0x80000000u);
}
__device__ __forceinline__ float fmax_decode(unsigned int k) {
  const unsigned int u = (k & 0x80000000u) ? (k ^ 0x80000000u) : ~k;
  return __uint_as_float(u);
}

__global__ __launch_bounds__(256)
void edge_score(const int* __restrict__ src, const int* __restrict__ dst, int E,
                int srcBase, int segOff, const bf16* __restrict__ qb,
                const bf16* __restrict__ kkb, const float* __restrict__ p_rel,
                int et, float* __restrict__ aOut,
                unsigned int* __restrict__ amaxEnc) {
  const int idx = blockIdx.x * 256 + threadIdx.x;
  if (idx >= E * 4) return;
  const int e = idx >> 2, h = idx & 3;
  const int seg = dst[e] + segOff;
  const bf16* qp = qb + (size_t)seg * 256 + h * 64;
  const bf16* kp = kkb + (size_t)(srcBase + src[e]) * 256 + h * 64;
  float acc = 0.f;
  #pragma unroll
  for (int i = 0; i < 64; ++i) acc += (float)qp[i] * (float)kp[i];
  acc *= p_rel[et * 4 + h] * 0.125f;             // 1/sqrt(64)
  aOut[(size_t)e * 4 + h] = acc;
  atomicMax(&amaxEnc[seg * 4 + h], fmax_encode(acc));
}

// ---------------------------------------------------------------------------
// Kernel 4: unnormalized softmax accumulate: esum += e, outAcc += e * vv[src]
// ---------------------------------------------------------------------------
__global__ __launch_bounds__(256)
void edge_accum(const int* __restrict__ src, const int* __restrict__ dst, int E,
                int srcBase, int segOff, const bf16* __restrict__ vvb,
                const float* __restrict__ aIn,
                const unsigned int* __restrict__ amaxEnc,
                float* __restrict__ esum, float* __restrict__ outAcc) {
  const int idx = blockIdx.x * 256 + threadIdx.x;
  if (idx >= E * 4) return;
  const int e = idx >> 2, h = idx & 3;
  const int seg = dst[e] + segOff;
  const float amax = fmax_decode(amaxEnc[seg * 4 + h]);
  const float w = expf(aIn[(size_t)e * 4 + h] - amax);
  atomicAdd(&esum[seg * 4 + h], w);
  const bf16* vp = vvb + (size_t)(srcBase + src[e]) * 256 + h * 64;
  float* op = outAcc + (size_t)seg * 256 + h * 64;
  #pragma unroll
  for (int i = 0; i < 64; ++i) atomicAdd(&op[i], w * (float)vp[i]);
}

// ---------------------------------------------------------------------------
// Kernel 5a: normalize + exact GELU -> bf16 buffer
// ---------------------------------------------------------------------------
__global__ __launch_bounds__(256)
void finalize_gelu(const float* __restrict__ outAcc, const float* __restrict__ esum,
                   bf16* __restrict__ hbuf, int total) {
  const int idx = blockIdx.x * 256 + threadIdx.x;
  if (idx >= total) return;
  const int node = idx >> 8, h = (idx >> 6) & 3;
  const float v = outAcc[idx] / (esum[node * 4 + h] + 1e-16f);
  const float g = 0.5f * v * (1.0f + erff(v * 0.70710678118654752f));
  hbuf[idx] = (bf16)g;
}

// ---------------------------------------------------------------------------
// Kernel 5b: out GEMM + bias + sigmoid-skip blend + ReLU -> d_out (fp32)
// A operand bf16 -> async global->LDS copy. grid = (ceil(M/128), 4)
// ---------------------------------------------------------------------------
__global__ __launch_bounds__(256)
void out_gemm(const bf16* __restrict__ Hin, const float* __restrict__ Wout,
              const float* __restrict__ bout, const float* __restrict__ skip,
              const float* __restrict__ Xres, int M, float* __restrict__ outP) {
  __shared__ bf16 As[2][128 * 32];
  __shared__ bf16 Bs[2][64 * 32];
  const int tid = threadIdx.x, lane = tid & 31, wave = tid >> 5;
  const int wm = wave >> 1, wn = wave & 1;
  const int row0 = blockIdx.x * 128, col0 = blockIdx.y * 64;

  const int bn = tid & 63, bk0 = (tid >> 6) * 8;
  float breg[8];

  auto asyncA = [&](int kb, int buf) {
    #pragma unroll
    for (int q = 0; q < 2; ++q) {
      const int u = tid + q * 256;
      const int r = u >> 2, k8 = (u & 3) * 8;
      const int grow = row0 + r;
      if (grow < M)
        async_copy_b128(Hin + (size_t)grow * 256 + kb + k8,
                        &As[buf][r * 32 + k8]);
    }
  };
  auto loadB = [&](int kb) {
    #pragma unroll
    for (int u = 0; u < 8; ++u)
      breg[u] = Wout[(size_t)(kb + bk0 + u) * 256 + col0 + bn];
  };
  auto storeB = [&](int buf) {
    bf16* brow = &Bs[buf][bn * 32 + bk0];
    #pragma unroll
    for (int u = 0; u < 8; ++u) brow[u] = (bf16)breg[u];
  };

  v8f a00 = {}, a01 = {}, a10 = {}, a11 = {};
  asyncA(0, 0);
  loadB(0);
  storeB(0);
  async_wait0();
  __syncthreads();
  constexpr int S = 256 / 32;
  #pragma unroll
  for (int s = 0; s < S; ++s) {
    const int cur = s & 1;
    if (s + 1 < S) { asyncA((s + 1) * 32, 1 - cur); loadB((s + 1) * 32); }
    mma_step(As[cur], Bs[cur], wm, wn, lane, a00, a01, a10, a11);
    if (s + 1 < S) { storeB(1 - cur); async_wait0(); }
    __syncthreads();
  }

  const float al = 1.0f / (1.0f + expf(-skip[0]));
  const int nl = lane & 15, half = lane >> 4;
  const int colA = col0 + wn * 32 + nl;
  const int colB = colA + 16;
  const float bvA = bout[colA], bvB = bout[colB];
  #pragma unroll
  for (int r = 0; r < 8; ++r) {
    const int m0 = row0 + wm * 32 + r + 8 * half;
    const int m1 = m0 + 16;
    if (m0 < M) {
      float oA = al * (a00[r] + bvA) + (1.0f - al) * Xres[(size_t)m0 * 256 + colA];
      float oB = al * (a01[r] + bvB) + (1.0f - al) * Xres[(size_t)m0 * 256 + colB];
      outP[(size_t)m0 * 256 + colA] = oA > 0.f ? oA : 0.f;
      outP[(size_t)m0 * 256 + colB] = oB > 0.f ? oB : 0.f;
    }
    if (m1 < M) {
      float oA = al * (a10[r] + bvA) + (1.0f - al) * Xres[(size_t)m1 * 256 + colA];
      float oB = al * (a11[r] + bvB) + (1.0f - al) * Xres[(size_t)m1 * 256 + colB];
      outP[(size_t)m1 * 256 + colA] = oA > 0.f ? oA : 0.f;
      outP[(size_t)m1 * 256 + colB] = oB > 0.f ? oB : 0.f;
    }
  }
}

// ---------------------------------------------------------------------------
// Kernel 0: init accumulators
// ---------------------------------------------------------------------------
__global__ __launch_bounds__(256)
void init_ws(float* __restrict__ esum, unsigned int* __restrict__ amaxEnc,
             float* __restrict__ outAcc, int totalAcc, int totalSeg) {
  const int idx = blockIdx.x * 256 + threadIdx.x;
  if (idx < totalAcc) outAcc[idx] = 0.f;
  if (idx < totalSeg) { esum[idx] = 0.f; amaxEnc[idx] = 0u; }
}

// ---------------------------------------------------------------------------
// Host launcher
// ---------------------------------------------------------------------------
extern "C" void kernel_launch(void* const* d_in, const int* in_sizes, int n_in,
                              void* d_out, int out_size, void* d_ws, size_t ws_size,
                              hipStream_t stream) {
  (void)in_sizes; (void)n_in; (void)out_size; (void)ws_size;
  const float* x[4]    = {(const float*)d_in[0], (const float*)d_in[3],
                          (const float*)d_in[6], (const float*)d_in[9]};
  const float* Wkqv[4] = {(const float*)d_in[1], (const float*)d_in[4],
                          (const float*)d_in[7], (const float*)d_in[10]};
  const float* bkqv[4] = {(const float*)d_in[2], (const float*)d_in[5],
                          (const float*)d_in[8], (const float*)d_in[11]};
  const float* Wout_u = (const float*)d_in[12];
  const float* bout_u = (const float*)d_in[13];
  const float* skip_u = (const float*)d_in[14];
  const float* Wout_i = (const float*)d_in[15];
  const float* bout_i = (const float*)d_in[16];
  const float* skip_i = (const float*)d_in[17];
  const float* Wk_rel = (const float*)d_in[18];
  const float* Wv_rel = (const float*)d_in[19];
  const float* p_rel  = (const float*)d_in[20];
  const int* srcs[4]  = {(const int*)d_in[21], (const int*)d_in[23],
                         (const int*)d_in[25], (const int*)d_in[27]};
  const int* dsts[4]  = {(const int*)d_in[22], (const int*)d_in[24],
                         (const int*)d_in[26], (const int*)d_in[28]};

  const int Nn[4]       = {20000, 50000, 50000, 50000};  // user,item,taste,image
  const int nodeBase[4] = {0, 20000, 70000, 120000};
  const int etOfType[4] = {2, 3, 0, 1};
  const int Ecnt[4]     = {50000, 50000, 200000, 200000};
  const int srcBase[4]  = {70000, 120000, 0, 20000};
  const int segOff[4]   = {20000, 20000, 20000, 0};
  const size_t eOff[4]  = {0, 50000, 100000, 300000};

  char* ws = (char*)d_ws;
  size_t off = 0;
  auto take = [&](size_t bytes) {
    char* p = ws + off;
    off = (off + bytes + 255) & ~(size_t)255;
    return p;
  };
  bf16* q_bf    = (bf16*)take((size_t)70000 * 256 * 2);
  bf16* k_bf    = (bf16*)take((size_t)170000 * 256 * 2);
  bf16* v_bf    = (bf16*)take((size_t)170000 * 256 * 2);
  bf16* kk_bf   = (bf16*)take((size_t)170000 * 256 * 2);
  bf16* vv_bf   = (bf16*)take((size_t)170000 * 256 * 2);
  float* a_buf  = (float*)take((size_t)500000 * 4 * 4);
  unsigned int* amaxE = (unsigned int*)take((size_t)70000 * 4 * 4);
  float* esum   = (float*)take((size_t)70000 * 4 * 4);
  float* outAcc = (float*)take((size_t)70000 * 256 * 4);
  bf16* hbuf    = (bf16*)take((size_t)70000 * 256 * 2);

  const int totalAcc = 70000 * 256;
  init_ws<<<dim3((totalAcc + 255) / 256), dim3(256), 0, stream>>>(
      esum, amaxE, outAcc, totalAcc, 70000 * 4);

  // 1) KQV projections (q stored only for user/item at segment rows)
  for (int t = 0; t < 4; ++t) {
    bf16* qPtr = nullptr;
    if (t == 0) qPtr = q_bf;
    if (t == 1) qPtr = q_bf + (size_t)20000 * 256;
    dim3 grid((Nn[t] + 127) / 128, 12);
    kqv_gemm<<<grid, dim3(256), 0, stream>>>(
        x[t], Wkqv[t], bkqv[t], Nn[t],
        k_bf + (size_t)nodeBase[t] * 256, qPtr,
        v_bf + (size_t)nodeBase[t] * 256);
  }

  // 2) relation transforms: k -> kk, v -> vv
  for (int t = 0; t < 4; ++t) {
    dim3 grid((Nn[t] + 127) / 128, 4);
    rel_gemm<<<grid, dim3(256), 0, stream>>>(
        k_bf + (size_t)nodeBase[t] * 256, Wk_rel, Nn[t], etOfType[t],
        kk_bf + (size_t)nodeBase[t] * 256);
    rel_gemm<<<grid, dim3(256), 0, stream>>>(
        v_bf + (size_t)nodeBase[t] * 256, Wv_rel, Nn[t], etOfType[t],
        vv_bf + (size_t)nodeBase[t] * 256);
  }

  // 3) edge scores + segment max
  for (int i = 0; i < 4; ++i) {
    dim3 grid(((size_t)Ecnt[i] * 4 + 255) / 256);
    edge_score<<<grid, dim3(256), 0, stream>>>(
        srcs[i], dsts[i], Ecnt[i], srcBase[i], segOff[i],
        q_bf, kk_bf, p_rel, i, a_buf + eOff[i] * 4, amaxE);
  }

  // 4) softmax-weighted accumulation
  for (int i = 0; i < 4; ++i) {
    dim3 grid(((size_t)Ecnt[i] * 4 + 255) / 256);
    edge_accum<<<grid, dim3(256), 0, stream>>>(
        srcs[i], dsts[i], Ecnt[i], srcBase[i], segOff[i],
        vv_bf, a_buf + eOff[i] * 4, amaxE, esum, outAcc);
  }

  // 5) normalize + GELU
  finalize_gelu<<<dim3((totalAcc + 255) / 256), dim3(256), 0, stream>>>(
      outAcc, esum, hbuf, totalAcc);

  // 6) output projections + skip + relu
  float* outF = (float*)d_out;
  {
    dim3 grid((20000 + 127) / 128, 4);
    out_gemm<<<grid, dim3(256), 0, stream>>>(
        hbuf, Wout_u, bout_u, skip_u, x[0], 20000, outF);
  }
  {
    dim3 grid((50000 + 127) / 128, 4);
    out_gemm<<<grid, dim3(256), 0, stream>>>(
        hbuf + (size_t)20000 * 256, Wout_i, bout_i, skip_i, x[1], 50000,
        outF + (size_t)20000 * 256);
  }
}
